// UMT_MVSNet_V2_46797963657722
// MI455X (gfx1250) — matrix operand order repaired and checked by reading, not compiled, not run on hardware
//
#include <hip/hip_runtime.h>
#include <hip/hip_bf16.h>
#include <math.h>

// ---------------------------------------------------------------------------
// Problem constants (from reference)
// ---------------------------------------------------------------------------
#define BB    2
#define VV    3
#define DD    32
#define HH    128
#define WW    128
#define HID   768
#define NHH   12
#define PP    16
#define NTOK  64            // (H/P)*(W/P)
#define HWPX  (HH*WW)       // 16384
#define BDN   (BB*DD)       // 64
#define M1    (BDN*NTOK)    // 4096 token rows in feat block
#define NSRC  (VV-1)

typedef _Float16 h8v  __attribute__((ext_vector_type(8)));
typedef _Float16 h16v __attribute__((ext_vector_type(16)));
typedef float    f8v  __attribute__((ext_vector_type(8)));

// ---------------------------------------------------------------------------
// Helpers
// ---------------------------------------------------------------------------
__device__ __forceinline__ float gelu_tanh(float x) {
    const float c0 = 0.7978845608028654f;  // sqrt(2/pi)
    return 0.5f * x * (1.0f + tanhf(c0 * (x + 0.044715f * x * x * x)));
}

__device__ void invert4x4(const float* m, float* inv) {
    inv[0]  =  m[5]*m[10]*m[15] - m[5]*m[11]*m[14] - m[9]*m[6]*m[15] + m[9]*m[7]*m[14] + m[13]*m[6]*m[11] - m[13]*m[7]*m[10];
    inv[4]  = -m[4]*m[10]*m[15] + m[4]*m[11]*m[14] + m[8]*m[6]*m[15] - m[8]*m[7]*m[14] - m[12]*m[6]*m[11] + m[12]*m[7]*m[10];
    inv[8]  =  m[4]*m[9]*m[15]  - m[4]*m[11]*m[13] - m[8]*m[5]*m[15] + m[8]*m[7]*m[13] + m[12]*m[5]*m[11] - m[12]*m[7]*m[9];
    inv[12] = -m[4]*m[9]*m[14]  + m[4]*m[10]*m[13] + m[8]*m[5]*m[14] - m[8]*m[6]*m[13] - m[12]*m[5]*m[10] + m[12]*m[6]*m[9];
    inv[1]  = -m[1]*m[10]*m[15] + m[1]*m[11]*m[14] + m[9]*m[2]*m[15] - m[9]*m[3]*m[14] - m[13]*m[2]*m[11] + m[13]*m[3]*m[10];
    inv[5]  =  m[0]*m[10]*m[15] - m[0]*m[11]*m[14] - m[8]*m[2]*m[15] + m[8]*m[3]*m[14] + m[12]*m[2]*m[11] - m[12]*m[3]*m[10];
    inv[9]  = -m[0]*m[9]*m[15]  + m[0]*m[11]*m[13] + m[8]*m[1]*m[15] - m[8]*m[3]*m[13] - m[12]*m[1]*m[11] + m[12]*m[3]*m[9];
    inv[13] =  m[0]*m[9]*m[14]  - m[0]*m[10]*m[13] - m[8]*m[1]*m[14] + m[8]*m[2]*m[13] + m[12]*m[1]*m[10] - m[12]*m[2]*m[9];
    inv[2]  =  m[1]*m[6]*m[15]  - m[1]*m[7]*m[14]  - m[5]*m[2]*m[15] + m[5]*m[3]*m[14] + m[13]*m[2]*m[7]  - m[13]*m[3]*m[6];
    inv[6]  = -m[0]*m[6]*m[15]  + m[0]*m[7]*m[14]  + m[4]*m[2]*m[15] - m[4]*m[3]*m[14] - m[12]*m[2]*m[7]  + m[12]*m[3]*m[6];
    inv[10] =  m[0]*m[5]*m[15]  - m[0]*m[7]*m[13]  - m[4]*m[1]*m[15] + m[4]*m[3]*m[13] + m[12]*m[1]*m[7]  - m[12]*m[3]*m[5];
    inv[14] = -m[0]*m[5]*m[14]  + m[0]*m[6]*m[13]  + m[4]*m[1]*m[14] - m[4]*m[2]*m[13] - m[12]*m[1]*m[6]  + m[12]*m[2]*m[5];
    inv[3]  = -m[1]*m[6]*m[11]  + m[1]*m[7]*m[10]  + m[5]*m[2]*m[11] - m[5]*m[3]*m[10] - m[9]*m[2]*m[7]   + m[9]*m[3]*m[6];
    inv[7]  =  m[0]*m[6]*m[11]  - m[0]*m[7]*m[10]  - m[4]*m[2]*m[11] + m[4]*m[3]*m[10] + m[8]*m[2]*m[7]   - m[8]*m[3]*m[6];
    inv[11] = -m[0]*m[5]*m[11]  + m[0]*m[7]*m[9]   + m[4]*m[1]*m[11] - m[4]*m[3]*m[9]  - m[8]*m[1]*m[7]   + m[8]*m[3]*m[5];
    inv[15] =  m[0]*m[5]*m[10]  - m[0]*m[6]*m[9]   - m[4]*m[1]*m[10] + m[4]*m[2]*m[9]  + m[8]*m[1]*m[6]   - m[8]*m[2]*m[5];
    float det = m[0]*inv[0] + m[1]*inv[4] + m[2]*inv[8] + m[3]*inv[12];
    float id = 1.0f / det;
    for (int i = 0; i < 16; ++i) inv[i] *= id;
}

__device__ __forceinline__ h16v load_a_frag(const _Float16* arow, int k, int half) {
    h8v lo = *(const h8v*)(arow + k + (half << 4));
    h8v hi = *(const h8v*)(arow + k + (half << 4) + 8);
    h16v a;
#pragma unroll
    for (int i = 0; i < 8; ++i) { a[i] = lo[i]; a[i + 8] = hi[i]; }
    return a;
}

// ---------------------------------------------------------------------------
// WMMA GEMM: C[M][N] = A[M][K](f16) * Bt[N][K](f16)^T + bias + res, act
// One wave computes a 32(M) x 64(N) strip: 2 A frags x 4 B frags -> 8 WMMAs
// per 32-wide k-step (24B/lane loaded per WMMA vs 64B for 1x1 tiling).
// Requires M%32==0, N%64==0, K%32==0, blockDim=32.
// act: 0 = none, 1 = gelu(tanh approx)
// ---------------------------------------------------------------------------
__global__ void __launch_bounds__(32)
wmma_gemm_kernel(const _Float16* __restrict__ A, const _Float16* __restrict__ Bt,
                 float* __restrict__ C, const float* __restrict__ bias,
                 const float* __restrict__ res, int M, int N, int K, int act)
{
    const int lane  = threadIdx.x & 31;
    const int half  = lane >> 4;   // 0: lanes 0-15, 1: lanes 16-31
    const int l     = lane & 15;
    const int strpN = blockIdx.x;  // 64-wide
    const int strpM = blockIdx.y;  // 32-wide

    const _Float16* arow0 = A + (size_t)(strpM * 32 + l) * K;        // rows +0..15
    const _Float16* arow1 = arow0 + (size_t)16 * K;                  // rows +16..31
    const _Float16* brow0 = Bt + (size_t)(strpN * 64 + 0  + l) * K;
    const _Float16* brow1 = Bt + (size_t)(strpN * 64 + 16 + l) * K;
    const _Float16* brow2 = Bt + (size_t)(strpN * 64 + 32 + l) * K;
    const _Float16* brow3 = Bt + (size_t)(strpN * 64 + 48 + l) * K;

    f8v c00 = {}, c01 = {}, c02 = {}, c03 = {};
    f8v c10 = {}, c11 = {}, c12 = {}, c13 = {};

    for (int k = 0; k < K; k += 32) {
        // A fragments: lanes 0-15 hold K {k..k+15}; lanes 16-31 K {k+16..k+31}.
        h16v a0 = load_a_frag(arow0, k, half);
        h16v a1 = load_a_frag(arow1, k, half);
        // B fragments: lanes 0-15 hold K {k..k+15}; lanes 16-31 K {k+16..k+31}.
        h16v b0 = *(const h16v*)(brow0 + k + (half << 4));
        h16v b1 = *(const h16v*)(brow1 + k + (half << 4));
        h16v b2 = *(const h16v*)(brow2 + k + (half << 4));
        h16v b3 = *(const h16v*)(brow3 + k + (half << 4));
        c00 = __builtin_amdgcn_wmma_f32_16x16x32_f16(false, a0, false, b0, (short)0, c00, false, false);
        c01 = __builtin_amdgcn_wmma_f32_16x16x32_f16(false, a0, false, b1, (short)0, c01, false, false);
        c02 = __builtin_amdgcn_wmma_f32_16x16x32_f16(false, a0, false, b2, (short)0, c02, false, false);
        c03 = __builtin_amdgcn_wmma_f32_16x16x32_f16(false, a0, false, b3, (short)0, c03, false, false);
        c10 = __builtin_amdgcn_wmma_f32_16x16x32_f16(false, a1, false, b0, (short)0, c10, false, false);
        c11 = __builtin_amdgcn_wmma_f32_16x16x32_f16(false, a1, false, b1, (short)0, c11, false, false);
        c12 = __builtin_amdgcn_wmma_f32_16x16x32_f16(false, a1, false, b2, (short)0, c12, false, false);
        c13 = __builtin_amdgcn_wmma_f32_16x16x32_f16(false, a1, false, b3, (short)0, c13, false, false);
    }

    // C/D layout: lanes 0-15 -> rows +0..7 (VGPR i = row i), lanes 16-31 -> rows +8..15
    f8v* cc[2][4] = { { &c00, &c01, &c02, &c03 }, { &c10, &c11, &c12, &c13 } };
#pragma unroll
    for (int j = 0; j < 4; ++j) {
        const int col = strpN * 64 + j * 16 + l;
        const float bv = bias ? bias[col] : 0.0f;
#pragma unroll
        for (int i = 0; i < 2; ++i) {
            const int rbase = strpM * 32 + i * 16 + half * 8;
            f8v c = *cc[i][j];
#pragma unroll
            for (int r = 0; r < 8; ++r) {
                const int row = rbase + r;
                float v = c[r] + bv;
                if (res) v += res[(size_t)row * N + col];
                if (act == 1) v = gelu_tanh(v);
                C[(size_t)row * N + col] = v;
            }
        }
    }
}

// ---------------------------------------------------------------------------
// Homography precompute: rot(3x3)+trans(3) per (batch, src)
// ---------------------------------------------------------------------------
__global__ void proj_kernel(const float* __restrict__ projs, float* __restrict__ mats)
{
    int t = blockIdx.x * blockDim.x + threadIdx.x;
    if (t >= BB * NSRC) return;
    int b = t / NSRC, s = t % NSRC;
    const float* ref = projs + ((size_t)b * VV + 0) * 16;
    const float* src = projs + ((size_t)b * VV + (s + 1)) * 16;
    float inv[16];
    invert4x4(ref, inv);
    float Pm[16];
    for (int i = 0; i < 4; ++i)
        for (int j = 0; j < 4; ++j) {
            float acc = 0.f;
            for (int k = 0; k < 4; ++k) acc += src[i * 4 + k] * inv[k * 4 + j];
            Pm[i * 4 + j] = acc;
        }
    float* o = mats + t * 12;
    for (int i = 0; i < 3; ++i)
        for (int j = 0; j < 3; ++j) o[i * 3 + j] = Pm[i * 4 + j];
    for (int i = 0; i < 3; ++i) o[9 + i] = Pm[i * 4 + 3];
}

// ---------------------------------------------------------------------------
// Warp + squared diff against reference view
// ---------------------------------------------------------------------------
__device__ __forceinline__ float tap(const float* ch, int yi, int xi) {
    bool valid = (xi >= 0) && (xi <= WW - 1) && (yi >= 0) && (yi <= HH - 1);
    int xc = min(max(xi, 0), WW - 1);
    int yc = min(max(yi, 0), HH - 1);
    return valid ? ch[yc * WW + xc] : 0.0f;
}

__global__ void warp_diff_kernel(const float* __restrict__ imgs, const float* __restrict__ mats,
                                 const float* __restrict__ depth_values,
                                 float* __restrict__ diff, int s)
{
    int idx = blockIdx.x * blockDim.x + threadIdx.x;
    if (idx >= BDN * HWPX) return;
    int p = idx % HWPX, n = idx / HWPX;
    int d = n % DD, b = n / DD;
    int x = p % WW, y = p / WW;

    const float* m = mats + ((size_t)b * NSRC + s) * 12;
    float fx = (float)x, fy = (float)y;
    float rx = m[0] * fx + m[1] * fy + m[2];
    float ry = m[3] * fx + m[4] * fy + m[5];
    float rz = m[6] * fx + m[7] * fy + m[8];
    float dep = depth_values[b * DD + d];
    float px = rx * dep + m[9], py = ry * dep + m[10], pz = rz * dep + m[11];
    float z = (fabsf(pz) < 1e-6f) ? 1e-6f : pz;
    float gx = px / z, gy = py / z;

    float x0f = floorf(gx), y0f = floorf(gy);
    float wx = gx - x0f, wy = gy - y0f;
    int x0 = (int)x0f, y0 = (int)y0f;

    const float* src = imgs + ((size_t)b * VV + (s + 1)) * 3 * HWPX;
    const float* ref = imgs + ((size_t)b * VV) * 3 * HWPX;
    for (int c = 0; c < 3; ++c) {
        const float* ch = src + (size_t)c * HWPX;
        float sval = tap(ch, y0, x0) * (1 - wx) * (1 - wy)
                   + tap(ch, y0, x0 + 1) * wx * (1 - wy)
                   + tap(ch, y0 + 1, x0) * (1 - wx) * wy
                   + tap(ch, y0 + 1, x0 + 1) * wx * wy;
        float dl = sval - ref[(size_t)c * HWPX + p];
        diff[((size_t)n * 3 + c) * HWPX + p] = dl * dl;
    }
}

// ---------------------------------------------------------------------------
// 3x3 SAME conv, act: 0=relu, 1=sigmoid
// ---------------------------------------------------------------------------
__global__ void conv3x3_kernel(const float* __restrict__ in, const float* __restrict__ w,
                               const float* __restrict__ bias, float* __restrict__ out,
                               int Nimg, int Ci, int Co, int act)
{
    long long idx = (long long)blockIdx.x * blockDim.x + threadIdx.x;
    long long total = (long long)Nimg * Co * HWPX;
    if (idx >= total) return;
    int p = (int)(idx % HWPX);
    long long rest = idx / HWPX;
    int co = (int)(rest % Co);
    int n  = (int)(rest / Co);
    int x = p % WW, y = p / WW;

    float acc = bias[co];
    for (int ci = 0; ci < Ci; ++ci) {
        const float* chan = in + ((size_t)n * Ci + ci) * HWPX;
        const float* wk = w + ((size_t)co * Ci + ci) * 9;
        for (int ky = -1; ky <= 1; ++ky) {
            int yy = y + ky;
            if (yy < 0 || yy >= HH) continue;
            for (int kx = -1; kx <= 1; ++kx) {
                int xx = x + kx;
                if (xx < 0 || xx >= WW) continue;
                acc += chan[yy * WW + xx] * wk[(ky + 1) * 3 + (kx + 1)];
            }
        }
    }
    if (act == 0) acc = fmaxf(acc, 0.0f);
    else          acc = 1.0f / (1.0f + expf(-acc));
    out[idx] = acc;
}

// acc (+)= (rw + 1) * diff
__global__ void accum_kernel(const float* __restrict__ diff, const float* __restrict__ rw,
                             float* __restrict__ acc, int first)
{
    long long idx = (long long)blockIdx.x * blockDim.x + threadIdx.x;
    if (idx >= (long long)BDN * 3 * HWPX) return;
    int p = (int)(idx % HWPX);
    long long rest = idx / HWPX;
    int n = (int)(rest / 3);
    float t = (rw[(size_t)n * HWPX + p] + 1.0f) * diff[idx];
    acc[idx] = first ? t : acc[idx] + t;
}

// im2col of vol (= acc / NSRC) into f16 token matrix [M1][HID]
__global__ void im2col_kernel(const float* __restrict__ acc, _Float16* __restrict__ A)
{
    long long idx = (long long)blockIdx.x * blockDim.x + threadIdx.x;
    if (idx >= (long long)M1 * HID) return;
    int k = (int)(idx % HID);
    int row = (int)(idx / HID);
    int n = row / NTOK, t = row % NTOK;
    int ox = t % (WW / PP), oy = t / (WW / PP);
    int c = k / (PP * PP);
    int r = k % (PP * PP);
    int py = r / PP, px = r % PP;
    float v = acc[((size_t)n * 3 + c) * HWPX + (oy * PP + py) * WW + (ox * PP + px)]
              * (1.0f / NSRC);
    A[idx] = (_Float16)v;
}

// W f32 [Kd][Nd] -> Wt f16 [Nd][Kd]
__global__ void wt_kernel(const float* __restrict__ Wsrc, _Float16* __restrict__ Wt,
                          int Kd, int Nd)
{
    long long idx = (long long)blockIdx.x * blockDim.x + threadIdx.x;
    if (idx >= (long long)Kd * Nd) return;
    int n = (int)(idx % Nd);
    int k = (int)(idx / Nd);
    Wt[(size_t)n * Kd + k] = (_Float16)Wsrc[idx];
}

// plain f32 -> f16 copy
__global__ void cvt_kernel(const float* __restrict__ src, _Float16* __restrict__ dst,
                           long long n)
{
    long long idx = (long long)blockIdx.x * blockDim.x + threadIdx.x;
    if (idx < n) dst[idx] = (_Float16)src[idx];
}

// x[row][c] += pos[row % NTOK][c]
__global__ void add_pos_kernel(float* __restrict__ x, const float* __restrict__ pos)
{
    long long idx = (long long)blockIdx.x * blockDim.x + threadIdx.x;
    if (idx >= (long long)M1 * HID) return;
    int c = (int)(idx % HID);
    int row = (int)(idx / HID);
    x[idx] += pos[(size_t)(row % NTOK) * HID + c];
}

// layernorm over HID, f32 in -> f16 out, one block (256 thr) per row
__global__ void __launch_bounds__(256)
layernorm_f16_kernel(const float* __restrict__ xin, const float* __restrict__ g,
                     const float* __restrict__ bta, _Float16* __restrict__ outh)
{
    int row = blockIdx.x;
    int tid = threadIdx.x;
    const float* xr = xin + (size_t)row * HID;
    __shared__ float red[256];

    float s = 0.f;
    for (int c = tid; c < HID; c += 256) s += xr[c];
    red[tid] = s; __syncthreads();
    for (int st = 128; st > 0; st >>= 1) { if (tid < st) red[tid] += red[tid + st]; __syncthreads(); }
    float mean = red[0] * (1.0f / HID);
    __syncthreads();

    float v = 0.f;
    for (int c = tid; c < HID; c += 256) { float d = xr[c] - mean; v += d * d; }
    red[tid] = v; __syncthreads();
    for (int st = 128; st > 0; st >>= 1) { if (tid < st) red[tid] += red[tid + st]; __syncthreads(); }
    float rstd = rsqrtf(red[0] * (1.0f / HID) + 1e-5f);
    __syncthreads();

    _Float16* orow = outh + (size_t)row * HID;
    for (int c = tid; c < HID; c += 256)
        orow[c] = (_Float16)((xr[c] - mean) * rstd * g[c] + bta[c]);
}

// attention: thread per (bn, head, q); hd = 64; Ntok <= 64
__global__ void attention_kernel(const float* __restrict__ qkv, float* __restrict__ outb,
                                 int Bn, int Ntok)
{
    int t = blockIdx.x * blockDim.x + threadIdx.x;
    if (t >= Bn * NHH * Ntok) return;
    int q = t % Ntok;
    int rest = t / Ntok;
    int h = rest % NHH;
    int bn = rest / NHH;
    const int hd = HID / NHH;                 // 64
    const size_t rowstride = 3 * HID;

    const float* qrow = qkv + ((size_t)(bn * Ntok + q)) * rowstride + h * hd;
    float s[64];
    float mx = -1e30f;
    for (int kk = 0; kk < Ntok; ++kk) {
        const float* krow = qkv + ((size_t)(bn * Ntok + kk)) * rowstride + HID + h * hd;
        float dot = 0.f;
        for (int dd = 0; dd < hd; ++dd) dot += qrow[dd] * krow[dd];
        dot *= 0.125f;                        // hd^-0.5
        s[kk] = dot;
        mx = fmaxf(mx, dot);
    }
    float sum = 0.f;
    for (int kk = 0; kk < Ntok; ++kk) { s[kk] = expf(s[kk] - mx); sum += s[kk]; }
    float inv = 1.0f / sum;

    float o[64];
    for (int dd = 0; dd < hd; ++dd) o[dd] = 0.f;
    for (int kk = 0; kk < Ntok; ++kk) {
        float wgt = s[kk] * inv;
        const float* vrow = qkv + ((size_t)(bn * Ntok + kk)) * rowstride + 2 * HID + h * hd;
        for (int dd = 0; dd < hd; ++dd) o[dd] += wgt * vrow[dd];
    }
    float* orow = outb + ((size_t)(bn * Ntok + q)) * HID + h * hd;
    for (int dd = 0; dd < hd; ++dd) orow[dd] = o[dd];
}

// mean over NTOK tokens -> feat [BDN][HID]
__global__ void mean_tokens_kernel(const float* __restrict__ xin, float* __restrict__ feat)
{
    int idx = blockIdx.x * blockDim.x + threadIdx.x;
    if (idx >= BDN * HID) return;
    int c = idx % HID, n = idx / HID;
    float s = 0.f;
    for (int t = 0; t < NTOK; ++t) s += xin[((size_t)n * NTOK + t) * HID + c];
    feat[idx] = s * (1.0f / NTOK);
}

// softmax over D + depth regression + 4-tap confidence
__global__ void softmax_depth_conf_kernel(const float* __restrict__ dec,
                                          const float* __restrict__ depth_values,
                                          float* __restrict__ depth_out,
                                          float* __restrict__ prob_out,
                                          float* __restrict__ conf_out)
{
    int idx = blockIdx.x * blockDim.x + threadIdx.x;
    if (idx >= BB * HWPX) return;
    int p = idx % HWPX, b = idx / HWPX;

    float mx = -1e30f;
    for (int d = 0; d < DD; ++d)
        mx = fmaxf(mx, dec[((size_t)(b * DD + d)) * HWPX + p]);
    float e[DD];
    float sum = 0.f;
    for (int d = 0; d < DD; ++d) {
        e[d] = expf(dec[((size_t)(b * DD + d)) * HWPX + p] - mx);
        sum += e[d];
    }
    float inv = 1.0f / sum;
    float dep = 0.f, ei = 0.f;
    for (int d = 0; d < DD; ++d) {
        float pr = e[d] * inv;
        e[d] = pr;
        prob_out[((size_t)(b * DD + d)) * HWPX + p] = pr;
        dep += pr * depth_values[b * DD + d];
        ei  += pr * (float)d;
    }
    depth_out[idx] = dep;
    int di = (int)ei;
    di = min(max(di, 0), DD - 1);
    float c4 = 0.f;
    for (int j = di - 1; j <= di + 2; ++j)
        if (j >= 0 && j < DD) c4 += e[j];
    conf_out[idx] = c4;
}

// ---------------------------------------------------------------------------
// Host orchestration
// ---------------------------------------------------------------------------
static inline dim3 grid1d(long long n, int blk = 256) {
    return dim3((unsigned)((n + blk - 1) / blk));
}

extern "C" void kernel_launch(void* const* d_in, const int* in_sizes, int n_in,
                              void* d_out, int out_size, void* d_ws, size_t ws_size,
                              hipStream_t stream)
{
    // Input order: recursive dict-insertion-order flattening of setup_inputs().
    const float* imgs    = (const float*)d_in[0];
    const float* projs   = (const float*)d_in[1];
    const float* depthsv = (const float*)d_in[2];
    const float* g_w1 = (const float*)d_in[3],  *g_b1 = (const float*)d_in[4];
    const float* g_w2 = (const float*)d_in[5],  *g_b2 = (const float*)d_in[6];
    const float* g_w3 = (const float*)d_in[7],  *g_b3 = (const float*)d_in[8];
    const float* patch_w = (const float*)d_in[9], *patch_b = (const float*)d_in[10];
    const float* pos = (const float*)d_in[11];
    // feat block
    const float* f_ln1g = (const float*)d_in[12], *f_ln1b = (const float*)d_in[13];
    const float* f_qkvw = (const float*)d_in[14], *f_qkvb = (const float*)d_in[15];
    const float* f_pw   = (const float*)d_in[16], *f_pb   = (const float*)d_in[17];
    const float* f_ln2g = (const float*)d_in[18], *f_ln2b = (const float*)d_in[19];
    const float* f_w1   = (const float*)d_in[20], *f_b1   = (const float*)d_in[21];
    const float* f_w2   = (const float*)d_in[22], *f_b2   = (const float*)d_in[23];
    // cost block
    const float* c_ln1g = (const float*)d_in[24], *c_ln1b = (const float*)d_in[25];
    const float* c_qkvw = (const float*)d_in[26], *c_qkvb = (const float*)d_in[27];
    const float* c_pw   = (const float*)d_in[28], *c_pb   = (const float*)d_in[29];
    const float* c_ln2g = (const float*)d_in[30], *c_ln2b = (const float*)d_in[31];
    const float* c_w1   = (const float*)d_in[32], *c_b1   = (const float*)d_in[33];
    const float* c_w2   = (const float*)d_in[34], *c_b2   = (const float*)d_in[35];
    const float* dec_w  = (const float*)d_in[36], *dec_b  = (const float*)d_in[37];

    float* depth_out = (float*)d_out;                                   // 2*128*128
    float* prob_out  = depth_out + (size_t)BB * HWPX;                   // 2*32*128*128
    float* conf_out  = prob_out + (size_t)BB * DD * HWPX;               // 2*128*128

    // ---- workspace bump allocator (256B aligned offsets) ----
    char* ws = (char*)d_ws;
    size_t off = 0;
    auto alloc = [&](size_t bytes) -> char* {
        char* p = ws + off;
        off += (bytes + 255) & ~(size_t)255;
        return p;
    };
    float*     mats    = (float*)alloc(BB * NSRC * 12 * sizeof(float));
    float*     acc     = (float*)alloc((size_t)BDN * 3 * HWPX * sizeof(float));
    float*     xbuf    = (float*)alloc((size_t)M1 * HID * sizeof(float));   // tokens / x3
    float*     x2buf   = (float*)alloc((size_t)M1 * HID * sizeof(float));
    float*     feat64  = (float*)alloc((size_t)BDN * HID * sizeof(float));
    float*     cx2     = (float*)alloc((size_t)BDN * HID * sizeof(float));
    float*     cx3     = (float*)alloc((size_t)BDN * HID * sizeof(float));
    float*     decbuf  = (float*)alloc((size_t)BDN * HWPX * sizeof(float));
    // f16 weights (pre-transposed to [N][K])
    _Float16*  patch_wt = (_Float16*)alloc((size_t)HID * HID * 2);
    _Float16*  f_qkvwt  = (_Float16*)alloc((size_t)3 * HID * HID * 2);
    _Float16*  f_pwt    = (_Float16*)alloc((size_t)HID * HID * 2);
    _Float16*  f_w1t    = (_Float16*)alloc((size_t)4 * HID * HID * 2);
    _Float16*  f_w2t    = (_Float16*)alloc((size_t)4 * HID * HID * 2);
    _Float16*  c_qkvwt  = (_Float16*)alloc((size_t)3 * HID * HID * 2);
    _Float16*  c_pwt    = (_Float16*)alloc((size_t)HID * HID * 2);
    _Float16*  c_w1t    = (_Float16*)alloc((size_t)4 * HID * HID * 2);
    _Float16*  c_w2t    = (_Float16*)alloc((size_t)4 * HID * HID * 2);
    _Float16*  dec_wt   = (_Float16*)alloc((size_t)HWPX * HID * 2);
    // shared arena: stage-1 (warp/gate) scratch overlaps stage-2 (transformer)
    char* arena = ws + off;
    // stage-1 views
    float* diff = (float*)arena;
    float* h1   = diff + (size_t)BDN * 3 * HWPX;
    float* h2   = h1 + (size_t)BDN * 8 * HWPX;
    float* rw   = h2 + (size_t)BDN * 8 * HWPX;
    // stage-2 views (same arena)
    float*    qkv    = (float*)arena;
    float*    attn_o = qkv + (size_t)M1 * 3 * HID;
    float*    mlp    = attn_o + (size_t)M1 * HID;
    _Float16* a16    = (_Float16*)(mlp + (size_t)M1 * 4 * HID);
    _Float16* h16    = a16 + (size_t)M1 * HID;

    const dim3 blk(256);

    // ---- stage 0: weight conversion ----
    // patch_w (HID,3,P,P) flat is already [N=HID][K=768] -> plain convert
    cvt_kernel<<<grid1d((long long)HID * HID), blk, 0, stream>>>(patch_w, patch_wt, (long long)HID * HID);
    wt_kernel<<<grid1d((long long)HID * 3 * HID), blk, 0, stream>>>(f_qkvw, f_qkvwt, HID, 3 * HID);
    wt_kernel<<<grid1d((long long)HID * HID), blk, 0, stream>>>(f_pw, f_pwt, HID, HID);
    wt_kernel<<<grid1d((long long)HID * 4 * HID), blk, 0, stream>>>(f_w1, f_w1t, HID, 4 * HID);
    wt_kernel<<<grid1d((long long)4 * HID * HID), blk, 0, stream>>>(f_w2, f_w2t, 4 * HID, HID);
    wt_kernel<<<grid1d((long long)HID * 3 * HID), blk, 0, stream>>>(c_qkvw, c_qkvwt, HID, 3 * HID);
    wt_kernel<<<grid1d((long long)HID * HID), blk, 0, stream>>>(c_pw, c_pwt, HID, HID);
    wt_kernel<<<grid1d((long long)HID * 4 * HID), blk, 0, stream>>>(c_w1, c_w1t, HID, 4 * HID);
    wt_kernel<<<grid1d((long long)4 * HID * HID), blk, 0, stream>>>(c_w2, c_w2t, 4 * HID, HID);
    wt_kernel<<<grid1d((long long)HID * HWPX), blk, 0, stream>>>(dec_w, dec_wt, HID, HWPX);

    // ---- stage 1: warp + gatenet + cost-volume accumulation ----
    proj_kernel<<<dim3(1), dim3(32), 0, stream>>>(projs, mats);
    for (int s = 0; s < NSRC; ++s) {
        warp_diff_kernel<<<grid1d((long long)BDN * HWPX), blk, 0, stream>>>(imgs, mats, depthsv, diff, s);
        conv3x3_kernel<<<grid1d((long long)BDN * 8 * HWPX), blk, 0, stream>>>(diff, g_w1, g_b1, h1, BDN, 3, 8, 0);
        conv3x3_kernel<<<grid1d((long long)BDN * 8 * HWPX), blk, 0, stream>>>(h1, g_w2, g_b2, h2, BDN, 8, 8, 0);
        conv3x3_kernel<<<grid1d((long long)BDN * 1 * HWPX), blk, 0, stream>>>(h2, g_w3, g_b3, rw, BDN, 8, 1, 1);
        accum_kernel<<<grid1d((long long)BDN * 3 * HWPX), blk, 0, stream>>>(diff, rw, acc, s == 0 ? 1 : 0);
    }

    // ---- stage 2: patch embed (WMMA GEMM) ----
    im2col_kernel<<<grid1d((long long)M1 * HID), blk, 0, stream>>>(acc, a16);
    wmma_gemm_kernel<<<dim3(HID / 64, M1 / 32), dim3(32), 0, stream>>>(
        a16, patch_wt, xbuf, patch_b, nullptr, M1, HID, HID, 0);
    add_pos_kernel<<<grid1d((long long)M1 * HID), blk, 0, stream>>>(xbuf, pos);

    // ---- feat transformer block on [M1][HID] ----
    layernorm_f16_kernel<<<dim3(M1), dim3(256), 0, stream>>>(xbuf, f_ln1g, f_ln1b, a16);
    wmma_gemm_kernel<<<dim3(3 * HID / 64, M1 / 32), dim3(32), 0, stream>>>(
        a16, f_qkvwt, qkv, f_qkvb, nullptr, M1, 3 * HID, HID, 0);
    attention_kernel<<<grid1d((long long)BDN * NHH * NTOK), blk, 0, stream>>>(qkv, attn_o, BDN, NTOK);
    cvt_kernel<<<grid1d((long long)M1 * HID), blk, 0, stream>>>(attn_o, a16, (long long)M1 * HID);
    wmma_gemm_kernel<<<dim3(HID / 64, M1 / 32), dim3(32), 0, stream>>>(
        a16, f_pwt, x2buf, f_pb, xbuf, M1, HID, HID, 0);
    layernorm_f16_kernel<<<dim3(M1), dim3(256), 0, stream>>>(x2buf, f_ln2g, f_ln2b, a16);
    wmma_gemm_kernel<<<dim3(4 * HID / 64, M1 / 32), dim3(32), 0, stream>>>(
        a16, f_w1t, mlp, f_b1, nullptr, M1, 4 * HID, HID, 1);
    cvt_kernel<<<grid1d((long long)M1 * 4 * HID), blk, 0, stream>>>(mlp, h16, (long long)M1 * 4 * HID);
    wmma_gemm_kernel<<<dim3(HID / 64, M1 / 32), dim3(32), 0, stream>>>(
        h16, f_w2t, xbuf, f_b2, x2buf, M1, HID, 4 * HID, 0);

    // ---- token mean -> feat [BDN][HID] ----
    mean_tokens_kernel<<<grid1d((long long)BDN * HID), blk, 0, stream>>>(xbuf, feat64);

    // ---- cost transformer block on [BDN][HID] (Bn=2, Ntok=32) ----
    layernorm_f16_kernel<<<dim3(BDN), dim3(256), 0, stream>>>(feat64, c_ln1g, c_ln1b, a16);
    wmma_gemm_kernel<<<dim3(3 * HID / 64, BDN / 32), dim3(32), 0, stream>>>(
        a16, c_qkvwt, qkv, c_qkvb, nullptr, BDN, 3 * HID, HID, 0);
    attention_kernel<<<grid1d((long long)BB * NHH * DD), blk, 0, stream>>>(qkv, attn_o, BB, DD);
    cvt_kernel<<<grid1d((long long)BDN * HID), blk, 0, stream>>>(attn_o, a16, (long long)BDN * HID);
    wmma_gemm_kernel<<<dim3(HID / 64, BDN / 32), dim3(32), 0, stream>>>(
        a16, c_pwt, cx2, c_pb, feat64, BDN, HID, HID, 0);
    layernorm_f16_kernel<<<dim3(BDN), dim3(256), 0, stream>>>(cx2, c_ln2g, c_ln2b, a16);
    wmma_gemm_kernel<<<dim3(4 * HID / 64, BDN / 32), dim3(32), 0, stream>>>(
        a16, c_w1t, mlp, c_b1, nullptr, BDN, 4 * HID, HID, 1);
    cvt_kernel<<<grid1d((long long)BDN * 4 * HID), blk, 0, stream>>>(mlp, h16, (long long)BDN * 4 * HID);
    wmma_gemm_kernel<<<dim3(HID / 64, BDN / 32), dim3(32), 0, stream>>>(
        h16, c_w2t, cx3, c_b2, cx2, BDN, HID, 4 * HID, 0);

    // ---- decoder GEMM [64][16384] ----
    cvt_kernel<<<grid1d((long long)BDN * HID), blk, 0, stream>>>(cx3, a16, (long long)BDN * HID);
    wmma_gemm_kernel<<<dim3(HWPX / 64, BDN / 32), dim3(32), 0, stream>>>(
        a16, dec_wt, decbuf, dec_b, nullptr, BDN, HWPX, HID, 0);

    // ---- softmax over D, depth regression, confidence ----
    softmax_depth_conf_kernel<<<grid1d((long long)BB * HWPX), blk, 0, stream>>>(
        decbuf, depthsv, depth_out, prob_out, conf_out);

    (void)in_sizes; (void)n_in; (void)out_size; (void)ws_size;
}
